// HDRTransformer_66013647339550
// MI455X (gfx1250) — matrix operand order repaired and checked by reading, not compiled, not run on hardware
//
#include <hip/hip_runtime.h>
#include <hip/hip_bf16.h>
#include <math.h>

// ---- problem constants ----
#define HW      262144   // 512*512
#define C_DIM   96
#define NHEADS  6
#define HD      16
#define NTOK    64
#define MLPH    384

// ---- workspace layout (bytes) ----
#define X1_BYTES    100663296ULL          // 512*512*96 * 4B  (x1 in [H*W, 96] f32)
#define QKVW_OFF    (X1_BYTES)            // 288*96 f16 (B-fragment swizzled)
#define PROJW_OFF   (QKVW_OFF + 55296ULL) // 96*96 f16
#define FC1W_OFF    (PROJW_OFF + 18432ULL)// 384*96 f16
#define FC2W_OFF    (FC1W_OFF + 73728ULL) // 96*384 f16

typedef __attribute__((ext_vector_type(16))) _Float16 v16h;
typedef __attribute__((ext_vector_type(8)))  _Float16 v8h;
typedef __attribute__((ext_vector_type(8)))  float    v8f;

__device__ inline v8f wmma32(v16h a, v16h b, v8f c) {
  // D = A(16x32 f16) * B(32x16 f16) + C(16x16 f32)
  return __builtin_amdgcn_wmma_f32_16x16x32_f16(
      /*neg_a=*/false, a, /*neg_b=*/false, b,
      /*c_mod=*/(short)0, c, /*reuse_a=*/false, /*reuse_b=*/false);
}

__device__ inline v16h combine(v8h lo, v8h hi) {
  v16h a;
#pragma unroll
  for (int e = 0; e < 8; ++e) { a[e] = lo[e]; a[e + 8] = hi[e]; }
  return a;
}

// A fragment 16x32 from row-major LDS: lane m = l%16,
// halves: k = kha+0..7 (lo, 16B aligned) and k = kha+8..15 (hi), kha = 16*(lane/16).
__device__ inline v16h frag_A_lds(const _Float16* base, int row0, int stride,
                                  int col0, int lane) {
  const int m   = lane & 15;
  const int kha = (lane & 16) ? 16 : 0;
  const _Float16* p = base + (row0 + m) * stride + col0 + kha;
  return combine(*(const v8h*)p, *(const v8h*)(p + 8));
}

// A fragment with K=16 zero-padded to 32 (attention q)
__device__ inline v16h frag_A_k16(const _Float16* base, int row0, int stride,
                                  int col0, int lane) {
  const int m   = lane & 15;
  const _Float16* p = base + (row0 + m) * stride + col0;
  v8h z = {};
  if (lane & 16) return combine(z, z);
  return combine(*(const v8h*)p, *(const v8h*)(p + 8));
}

// B fragment from pre-swizzled weights: one aligned 32B load per lane
__device__ inline v16h frag_B_w(const _Float16* wf, int frag, int lane) {
  return ((const v16h*)wf)[frag * 32 + lane];
}

// shifted-window mask class (only boundary windows wy==63 / wx==63 mix regions)
__device__ inline int swin_cls(int wy, int wx, int tok) {
  const int i = tok >> 3, j = tok & 7;
  const int rh = (wy == 63) ? ((i < 4) ? 1 : 2) : 0;
  const int rw = (wx == 63) ? ((j < 4) ? 1 : 2) : 0;
  return rh * 3 + rw;
}

// ------------------------------------------------------------------
// weight f32 -> f16, swizzled into WMMA B-fragment layout:
// dst[frag][lane][e] = W[(tn*16 + lane%16)*Kdim + kt*32 + (lane&16?16:0)+e]
// frag = tn*nk + kt
// ------------------------------------------------------------------
__device__ inline void swizzle_one(const float* __restrict__ W,
                                   _Float16* __restrict__ dst,
                                   int i, int Kdim, int nk) {
  const int frag = i >> 9;          // /512
  const int r    = i & 511;
  const int ln   = r >> 4;
  const int e    = r & 15;
  const int n    = ln & 15;
  const int k    = ((ln & 16) ? 16 : 0) + e;
  const int tn   = frag / nk;
  const int kt   = frag % nk;
  dst[i] = (_Float16)W[(tn * 16 + n) * Kdim + kt * 32 + k];
}

__global__ void swin_prep_weights(const float* __restrict__ qkv_w,
                                  const float* __restrict__ proj_w,
                                  const float* __restrict__ fc1_w,
                                  const float* __restrict__ fc2_w,
                                  _Float16* qkvw, _Float16* projw,
                                  _Float16* fc1w, _Float16* fc2w) {
  const int i = blockIdx.x * blockDim.x + threadIdx.x;
  if (i < 27648) swizzle_one(qkv_w, qkvw, i, 96, 3);    // 18 x 3 frags
  if (i < 9216)  swizzle_one(proj_w, projw, i, 96, 3);  //  6 x 3 frags
  if (i < 36864) {
    swizzle_one(fc1_w, fc1w, i, 96, 3);                 // 24 x  3 frags
    swizzle_one(fc2_w, fc2w, i, 384, 12);               //  6 x 12 frags
  }
}

// ------------------------------------------------------------------
// Kernel 1: LN1 + shifted-window attention + proj + residual
// one workgroup (8 waves) per 8x8 window
// ------------------------------------------------------------------
__global__ __launch_bounds__(256)
void swin_attn_kernel(const float* __restrict__ x,
                      const float* __restrict__ n1w, const float* __restrict__ n1b,
                      const _Float16* __restrict__ qkvw, const float* __restrict__ qkv_b,
                      const float* __restrict__ rpb,
                      const _Float16* __restrict__ projw, const float* __restrict__ proj_b,
                      float* __restrict__ x1) {
  __shared__ __align__(32) unsigned char smem[61440];
  _Float16* qh = (_Float16*)(smem);            // [64][96]  12KB (row-major)
  _Float16* kh = (_Float16*)(smem + 12288);    // [64][96]  12KB (row-major)
  _Float16* vt = (_Float16*)(smem + 24576);    // [96][64]  12KB (transposed!)
  _Float16* xn = (_Float16*)(smem + 36864);    // [64][96]  12KB (phase A/B)
  _Float16* Pl = (_Float16*)(smem + 36864);    // [64][64]   8KB (phase C, reuses xn)
  _Float16* ao = (_Float16*)(smem + 49152);    // [64][96]  12KB

  const int tid  = threadIdx.x;
  const int wave = tid >> 5;
  const int lane = tid & 31;
  const int wy = blockIdx.x >> 6;
  const int wx = blockIdx.x & 63;

  // ---- Phase A: gather shifted pixels + LayerNorm1 -> xn (f16) ----
  // 4 lanes per token, 24 channels each, shfl-xor reduction
  {
    const int tok = tid >> 2;
    const int qd  = tid & 3;
    const int ti = tok >> 3, tj = tok & 7;
    const int ph = (wy * 8 + ti + 4) & 511;    // cyclic shift by -4 folded in
    const int pw = (wx * 8 + tj + 4) & 511;
    const int pix = ph * 512 + pw;
    const int c0 = qd * 24;
    float s = 0.f, ss = 0.f;
    for (int c = c0; c < c0 + 24; ++c) {
      const float t = x[c * HW + pix];
      s += t; ss += t * t;
    }
    s  += __shfl_xor(s, 1, 32);  ss += __shfl_xor(ss, 1, 32);
    s  += __shfl_xor(s, 2, 32);  ss += __shfl_xor(ss, 2, 32);
    const float mu = s * (1.0f / 96.0f);
    const float rs = rsqrtf(ss * (1.0f / 96.0f) - mu * mu + 1e-5f);
    for (int c = c0; c < c0 + 24; ++c) {
      const float t = (x[c * HW + pix] - mu) * rs * n1w[c] + n1b[c];
      xn[tok * 96 + c] = (_Float16)t;
    }
  }
  __syncthreads();

  // ---- Phase B: QKV GEMM  [64x96] @ [96x288] -> q,k (row), v (transposed) ----
  // 4 M-tiles x 18 N-tiles = 72 tiles, 9 per wave, K = 96 (3 wmma)
  for (int t = wave; t < 72; t += 8) {
    const int tm = t / 18, tn = t % 18;
    v8f acc = {};
    for (int kb = 0; kb < 96; kb += 32) {
      v16h a = frag_A_lds(xn, tm * 16, 96, kb, lane);
      v16h b = frag_B_w(qkvw, tn * 3 + (kb >> 5), lane);
      acc = wmma32(a, b, acc);
    }
    const int cm = (lane & 16) ? 8 : 0;
    const int cn = lane & 15;
    const int j = tn * 16 + cn;                // 0..287 : [q|k|v] x head x d
    const float bj = qkv_b[j];
    if (tn < 12) {                             // q or k (tile-uniform branch)
      _Float16* dst = (tn < 6) ? qh : kh;
      const int hd = j % 96;
#pragma unroll
      for (int r = 0; r < 8; ++r)
        dst[(tm * 16 + cm + r) * 96 + hd] = (_Float16)(acc[r] + bj);
    } else {                                   // v, stored transposed [96][64]
      const int hd = j - 192;
#pragma unroll
      for (int r = 0; r < 8; ++r)
        vt[hd * 64 + tm * 16 + cm + r] = (_Float16)(acc[r] + bj);
    }
  }
  __syncthreads();

  // ---- Phase C: per-head attention ----
  const float scale = 0.25f;                   // 16^-0.5
  for (int head = 0; head < NHEADS; ++head) {
    // S = (q * scale) @ k^T + bias + mask, 16 tiles (2 per wave), K=16 padded
    for (int t = wave; t < 16; t += 8) {
      const int tm = t >> 2, tn = t & 3;
      v16h a = frag_A_k16(qh, tm * 16, 96, head * HD, lane);
      // B[k=d][n=m] = kh[m*96 + head*16 + d]: contiguous 16 halves per lane
      v16h b = {};
      if (lane < 16)
        b = *(const v16h*)(kh + (tn * 16 + lane) * 96 + head * HD);
      v8f acc = {};
      acc = wmma32(a, b, acc);
      const int cm = (lane & 16) ? 8 : 0;
      const int cn = lane & 15;
      const int mtok = tn * 16 + cn;
      const int clsm = swin_cls(wy, wx, mtok);
      const int i2 = mtok >> 3, j2 = mtok & 7;
#pragma unroll
      for (int r = 0; r < 8; ++r) {
        const int ntok = tm * 16 + cm + r;
        const int i1 = ntok >> 3, j1 = ntok & 7;
        const int ridx = (i1 - i2 + 7) * 15 + (j1 - j2 + 7);
        float v = acc[r] * scale + rpb[ridx * NHEADS + head];
        if (swin_cls(wy, wx, ntok) != clsm) v -= 100.f;
        Pl[ntok * 64 + mtok] = (_Float16)v;
      }
    }
    __syncthreads();

    // softmax over each of the 64 rows
    if (tid < 64) {
      _Float16* pr = Pl + tid * 64;
      float mx = -1e30f;
      for (int m = 0; m < 64; ++m) mx = fmaxf(mx, (float)pr[m]);
      float sum = 0.f;
      for (int m = 0; m < 64; ++m) sum += __expf((float)pr[m] - mx);
      const float inv = 1.f / sum;
      for (int m = 0; m < 64; ++m)
        pr[m] = (_Float16)(__expf((float)pr[m] - mx) * inv);
    }
    __syncthreads();

    // out = P @ v : 4 M-tiles, N=16, K=64 (wave-uniform branch)
    if (wave < 4) {
      const int tm = wave;
      const int n  = lane & 15;
      const int ko = (lane & 16) ? 16 : 0;
      v8f acc = {};
      for (int kb = 0; kb < 64; kb += 32) {
        v16h a = frag_A_lds(Pl, tm * 16, 64, kb, lane);
        // B[k=m][n=d] = vt[(head*16+d)*64 + m]: contiguous 16 halves per lane
        v16h b = *(const v16h*)(vt + (head * HD + n) * 64 + kb + ko);
        acc = wmma32(a, b, acc);
      }
      const int cm = (lane & 16) ? 8 : 0;
#pragma unroll
      for (int r = 0; r < 8; ++r)
        ao[(tm * 16 + cm + r) * 96 + head * HD + n] = (_Float16)acc[r];
    }
    __syncthreads();
  }

  // ---- Phase D: proj GEMM + bias + shortcut, scatter x1[pix][c] ----
  // 4 x 6 = 24 tiles, 3 per wave, K = 96
  for (int t = wave; t < 24; t += 8) {
    const int tm = t / 6, tn = t % 6;
    v8f acc = {};
    for (int kb = 0; kb < 96; kb += 32) {
      v16h a = frag_A_lds(ao, tm * 16, 96, kb, lane);
      v16h b = frag_B_w(projw, tn * 3 + (kb >> 5), lane);
      acc = wmma32(a, b, acc);
    }
    const int cm = (lane & 16) ? 8 : 0;
    const int cn = lane & 15;
    const int o  = tn * 16 + cn;
    const float pb = proj_b[o];
#pragma unroll
    for (int r = 0; r < 8; ++r) {
      const int tok = tm * 16 + cm + r;
      const int ti = tok >> 3, tj = tok & 7;
      const int ph = (wy * 8 + ti + 4) & 511;  // token writes back to its pixel
      const int pw = (wx * 8 + tj + 4) & 511;
      const int pix = ph * 512 + pw;
      x1[pix * 96 + o] = acc[r] + pb + x[o * HW + pix];
    }
  }
}

// ------------------------------------------------------------------
// Kernel 2: LN2 + MLP (fc1 + GELU + fc2) + residual, emit [C,H,W]
// one workgroup per 64 consecutive tokens
// ------------------------------------------------------------------
__global__ __launch_bounds__(256)
void swin_mlp_kernel(const float* __restrict__ x1,
                     const float* __restrict__ n2w, const float* __restrict__ n2b,
                     const _Float16* __restrict__ fc1w, const float* __restrict__ fc1b,
                     const _Float16* __restrict__ fc2w, const float* __restrict__ fc2b,
                     float* __restrict__ out) {
  __shared__ __align__(32) unsigned char smem[61440];
  _Float16* hln = (_Float16*)(smem);           // [64][96]  12KB
  _Float16* hh  = (_Float16*)(smem + 12288);   // [64][384] 48KB

  const int tid  = threadIdx.x;
  const int wave = tid >> 5;
  const int lane = tid & 31;
  const int base = blockIdx.x * 64;

  // LayerNorm2 -> f16 (4 lanes per token, shfl reduction)
  {
    const int tok = tid >> 2;
    const int qd  = tid & 3;
    const int c0  = qd * 24;
    const float* xr = x1 + (size_t)(base + tok) * 96;
    float s = 0.f, ss = 0.f;
    for (int c = c0; c < c0 + 24; ++c) { const float t = xr[c]; s += t; ss += t * t; }
    s  += __shfl_xor(s, 1, 32);  ss += __shfl_xor(ss, 1, 32);
    s  += __shfl_xor(s, 2, 32);  ss += __shfl_xor(ss, 2, 32);
    const float mu = s * (1.0f / 96.0f);
    const float rs = rsqrtf(ss * (1.0f / 96.0f) - mu * mu + 1e-5f);
    for (int c = c0; c < c0 + 24; ++c)
      hln[tok * 96 + c] = (_Float16)((xr[c] - mu) * rs * n2w[c] + n2b[c]);
  }
  __syncthreads();

  // fc1: [64x96] @ [96x384], 4 x 24 = 96 tiles (12 per wave), + exact GELU
  for (int t = wave; t < 96; t += 8) {
    const int tm = t / 24, tn = t % 24;
    v8f acc = {};
    for (int kb = 0; kb < 96; kb += 32) {
      v16h a = frag_A_lds(hln, tm * 16, 96, kb, lane);
      v16h b = frag_B_w(fc1w, tn * 3 + (kb >> 5), lane);
      acc = wmma32(a, b, acc);
    }
    const int cm = (lane & 16) ? 8 : 0;
    const int cn = lane & 15;
    const int j  = tn * 16 + cn;
    const float bj = fc1b[j];
#pragma unroll
    for (int r = 0; r < 8; ++r) {
      const float v = acc[r] + bj;
      const float g = 0.5f * v * (1.f + erff(v * 0.70710678f));
      hh[(tm * 16 + cm + r) * 384 + j] = (_Float16)g;
    }
  }
  __syncthreads();

  // fc2: [64x384] @ [384x96], 4 x 6 = 24 tiles (3 per wave), K = 384 (12 wmma)
  for (int t = wave; t < 24; t += 8) {
    const int tm = t / 6, tn = t % 6;
    v8f acc = {};
    for (int kb = 0; kb < 384; kb += 32) {
      v16h a = frag_A_lds(hh, tm * 16, 384, kb, lane);
      v16h b = frag_B_w(fc2w, tn * 12 + (kb >> 5), lane);
      acc = wmma32(a, b, acc);
    }
    const int cm = (lane & 16) ? 8 : 0;
    const int cn = lane & 15;
    const int o  = tn * 16 + cn;
    const float bo = fc2b[o];
#pragma unroll
    for (int r = 0; r < 8; ++r) {
      const int row = base + tm * 16 + cm + r;           // flat pixel id
      const float v = acc[r] + bo + x1[(size_t)row * 96 + o];
      out[(size_t)o * HW + row] = v;                     // [C,H,W]
    }
  }
}

// ------------------------------------------------------------------
extern "C" void kernel_launch(void* const* d_in, const int* in_sizes, int n_in,
                              void* d_out, int out_size, void* d_ws, size_t ws_size,
                              hipStream_t stream) {
  (void)in_sizes; (void)n_in; (void)out_size; (void)ws_size;
  const float* x      = (const float*)d_in[0];
  const float* n1w    = (const float*)d_in[1];
  const float* n1b    = (const float*)d_in[2];
  const float* qkv_w  = (const float*)d_in[3];
  const float* qkv_b  = (const float*)d_in[4];
  const float* rpb    = (const float*)d_in[5];
  const float* proj_w = (const float*)d_in[6];
  const float* proj_b = (const float*)d_in[7];
  const float* n2w    = (const float*)d_in[8];
  const float* n2b    = (const float*)d_in[9];
  const float* fc1_w  = (const float*)d_in[10];
  const float* fc1_b  = (const float*)d_in[11];
  const float* fc2_w  = (const float*)d_in[12];
  const float* fc2_b  = (const float*)d_in[13];

  char* ws = (char*)d_ws;
  float*    x1    = (float*)ws;
  _Float16* qkvw  = (_Float16*)(ws + QKVW_OFF);
  _Float16* projw = (_Float16*)(ws + PROJW_OFF);
  _Float16* fc1w  = (_Float16*)(ws + FC1W_OFF);
  _Float16* fc2w  = (_Float16*)(ws + FC2W_OFF);
  float* out = (float*)d_out;

  swin_prep_weights<<<144, 256, 0, stream>>>(qkv_w, proj_w, fc1_w, fc2_w,
                                             qkvw, projw, fc1w, fc2w);
  swin_attn_kernel<<<4096, 256, 0, stream>>>(x, n1w, n1b, qkvw, qkv_b, rpb,
                                             projw, proj_b, x1);
  swin_mlp_kernel<<<4096, 256, 0, stream>>>(x1, n2w, n2b, fc1w, fc1_b,
                                            fc2w, fc2_b, out);
}
